// MAMBA2_2D_60232621359573
// MI455X (gfx1250) — matrix operand order repaired and checked by reading, not compiled, not run on hardware
//
#include <hip/hip_runtime.h>
#include <hip/hip_bf16.h>

typedef __bf16 bf16;
typedef __bf16 v16bf __attribute__((ext_vector_type(16)));
typedef __bf16 v8bf  __attribute__((ext_vector_type(8)));
typedef float  v8f   __attribute__((ext_vector_type(8)));
typedef int    v4i   __attribute__((ext_vector_type(4)));

#define M_TOK   32768      // 8*64*64 tokens
#define DM      256        // d_model
#define DIN     512        // d_inner
#define DSTATE  64
#define NH      8
#define HDIM    64
#define LSEQ    64
#define NSEQ    512        // sequences per direction
#define CONVCH  640        // d_inner + 2*d_state
#define DPROJ   1160       // 2*512 + 2*64 + 8
#define NPADIN  1216       // ceil(1160/64)*64
#define TK      32
#define ASTR    40         // LDS row stride (bf16 elems), padded

// ---- optional gfx1250 async global->LDS path (probe via __has_builtin) ----
#if defined(__has_builtin)
#  if __has_builtin(__builtin_amdgcn_global_load_async_to_lds_b128) && \
      __has_builtin(__builtin_amdgcn_s_wait_asynccnt)
#    define USE_ASYNC_LDS 1
#  endif
#endif
#ifndef USE_ASYNC_LDS
#  define USE_ASYNC_LDS 0
#endif

typedef __attribute__((address_space(1))) v4i as1_v4i;   // global (prints as __device__)
typedef __attribute__((address_space(3))) v4i as3_v4i;   // LDS    (prints as __shared__)

__device__ __forceinline__ bf16 f2bf(float f) {
    unsigned u = __builtin_bit_cast(unsigned, f);
    u = (u + 0x7FFFu + ((u >> 16) & 1u)) >> 16;
    unsigned short s = (unsigned short)u;
    return __builtin_bit_cast(bf16, s);
}

__device__ __forceinline__ v16bf frag_a(const bf16* s, int m, int half) {
    const v8bf* p0 = (const v8bf*)(s + m * ASTR + half * 8);
    const v8bf* p1 = (const v8bf*)(s + m * ASTR + 16 + half * 8);
    v8bf lo = *p0, hi = *p1;
    return __builtin_shufflevector(lo, hi, 0,1,2,3,4,5,6,7,8,9,10,11,12,13,14,15);
}

__device__ __forceinline__ v16bf frag_b(const bf16* s, int n, int half) {
    const v8bf* p0 = (const v8bf*)(s + n * ASTR + half * 8);
    const v8bf* p1 = (const v8bf*)(s + n * ASTR + 16 + half * 8);
    v8bf lo = *p0, hi = *p1;
    return __builtin_shufflevector(lo, hi, 0,1,2,3,4,5,6,7,8,9,10,11,12,13,14,15);
}

// C = A(M x K, bf16) * B(Nout x K, bf16)^T ; 64x64 tile per workgroup, 8 waves.
// flags: bit0 = bf16 output, bit1 = row permute (b,i,j)->(b,j,i), bit2 = add bias
__global__ __launch_bounds__(256) void gemm_bf16_wmma(
    const bf16* __restrict__ A, int lda,
    const bf16* __restrict__ B, int ldb,
    void* __restrict__ C, int ldc,
    int Nout, int K, int flags, int colOff, const float* __restrict__ bias)
{
    __shared__ __align__(16) bf16 As[64 * ASTR];
    __shared__ __align__(16) bf16 Bs[64 * ASTR];

    const int tid  = threadIdx.x;
    const int m0   = blockIdx.x * 64;
    const int n0   = blockIdx.y * 64;
    const int wave = tid >> 5, lane = tid & 31;
    const int tm   = wave & 3;            // M tile 0..3
    const int tn0  = (wave >> 2) << 1;    // N tiles {0,1} or {2,3}
    const int mlane = lane & 15, half = lane >> 4;

    const int r  = tid >> 2;              // staging row 0..63
    const int c8 = (tid & 3) << 3;        // staging col chunk 0/8/16/24

    v8f acc0 = {}; v8f acc1 = {};

    for (int k0 = 0; k0 < K; k0 += TK) {
#if USE_ASYNC_LDS
        __syncthreads();                  // previous iter's fragment reads done
        __builtin_amdgcn_global_load_async_to_lds_b128(
            (as1_v4i*)(unsigned long long)(const void*)(A + (size_t)(m0 + r) * lda + k0 + c8),
            (as3_v4i*)(As + r * ASTR + c8), 0, 0);
        __builtin_amdgcn_global_load_async_to_lds_b128(
            (as1_v4i*)(unsigned long long)(const void*)(B + (size_t)(n0 + r) * ldb + k0 + c8),
            (as3_v4i*)(Bs + r * ASTR + c8), 0, 0);
        __builtin_amdgcn_s_wait_asynccnt(0);
        __syncthreads();
#else
        const v8bf* ga = (const v8bf*)(A + (size_t)(m0 + r) * lda + k0 + c8);
        const v8bf* gb = (const v8bf*)(B + (size_t)(n0 + r) * ldb + k0 + c8);
        v8bf av = *ga;
        v8bf bv = *gb;
        if (k0 + TK < K) {
            __builtin_prefetch((const char*)(A + (size_t)(m0 + r) * lda + k0 + TK + c8), 0, 3);
            __builtin_prefetch((const char*)(B + (size_t)(n0 + r) * ldb + k0 + TK + c8), 0, 3);
        }
        __syncthreads();                  // previous iter's fragment reads done
        *(v8bf*)(As + r * ASTR + c8) = av;
        *(v8bf*)(Bs + r * ASTR + c8) = bv;
        __syncthreads();
#endif

        v16bf af = frag_a(As, tm * 16 + mlane, half);
        v16bf b0 = frag_b(Bs, tn0 * 16 + mlane, half);
        v16bf b1 = frag_b(Bs, (tn0 + 1) * 16 + mlane, half);
        acc0 = __builtin_amdgcn_wmma_f32_16x16x32_bf16(false, af, false, b0,
                                                       (short)0, acc0, false, false);
        acc1 = __builtin_amdgcn_wmma_f32_16x16x32_bf16(false, af, false, b1,
                                                       (short)0, acc1, false, false);
    }

    const bool obf  = flags & 1;
    const bool perm = flags & 2;
    const bool badd = flags & 4;
#pragma unroll
    for (int tni = 0; tni < 2; ++tni) {
        v8f acc = tni ? acc1 : acc0;
        int col = n0 + (tn0 + tni) * 16 + mlane;
        if (col < Nout) {
#pragma unroll
            for (int v = 0; v < 8; ++v) {
                int row = m0 + tm * 16 + half * 8 + v;
                int drow = row;
                if (perm) {                 // (b, i, j) -> (b, j, i), 64x64
                    int rr = row & 4095;
                    drow = (row & ~4095) | ((rr & 63) << 6) | (rr >> 6);
                }
                if (obf) {
                    ((bf16*)C)[(size_t)drow * ldc + colOff + col] = f2bf(acc[v]);
                } else {
                    float val = acc[v];
                    if (badd) val += bias[col];
                    ((float*)C)[(size_t)drow * ldc + col] = val;
                }
            }
        }
    }
}

// dst[i] = i < srcElems ? bf16(src[i]) : 0   (handles zero padding of weights)
__global__ void cvt_pad_bf16(const float* __restrict__ src, bf16* __restrict__ dst,
                             int srcElems, int dstElems) {
    int i = blockIdx.x * 256 + threadIdx.x;
    if (i >= dstElems) return;
    dst[i] = (i < srcElems) ? f2bf(src[i]) : f2bf(0.f);
}

// x (b,h,w,c) -> bf16 rows in (b,w,h) order
__global__ void cvt_xv(const float* __restrict__ x, bf16* __restrict__ dst) {
    int i = blockIdx.x * 256 + threadIdx.x;
    if (i >= M_TOK * DM) return;
    int c = i & 255;
    int t = i >> 8;
    int w = t & 63, h = (t >> 6) & 63, b = t >> 12;
    dst[(size_t)((b << 12) | (w << 6) | h) * DM + c] = f2bf(x[i]);
}

// fold the duplicated halves of fc_w: wcat(256 x 512)
__global__ void build_wcat(const float* __restrict__ fc_w, bf16* __restrict__ wcat) {
    int i = blockIdx.x * 256 + threadIdx.x;
    if (i >= 256 * 512) return;
    int n = i >> 9, k = i & 511;
    float v;
    if (k < 256) v = fc_w[n * 1024 + k]       + fc_w[n * 1024 + 256 + k];
    else         v = fc_w[n * 1024 + 256 + k] + fc_w[n * 1024 + 512 + k]; // 512+(k-256), 768+(k-256)
    wcat[i] = f2bf(v);
}

// causal depthwise conv (width 4) + bias + silu over xBC channels
__global__ __launch_bounds__(256) void conv_silu(
    const float* __restrict__ zx, const float* __restrict__ cw,
    const float* __restrict__ cb, float* __restrict__ xbc)
{
    int s = blockIdx.x;                         // sequence 0..511
    for (int cc = threadIdx.x; cc < CONVCH; cc += 256) {
        float w0 = cw[cc * 4 + 0], w1 = cw[cc * 4 + 1];
        float w2 = cw[cc * 4 + 2], w3 = cw[cc * 4 + 3];
        float bia = cb[cc];
        float x1 = 0.f, x2 = 0.f, x3 = 0.f;
        const float* src = zx + (size_t)s * LSEQ * DPROJ + DIN + cc;
        float* dstp = xbc + (size_t)s * LSEQ * CONVCH + cc;
        for (int t = 0; t < LSEQ; ++t) {
            float xv = src[(size_t)t * DPROJ];
            float o = bia + w0 * x3 + w1 * x2 + w2 * x1 + w3 * xv;
            x3 = x2; x2 = x1; x1 = xv;
            o = o / (1.f + __expf(-o));          // silu
            dstp[(size_t)t * CONVCH] = o;
        }
    }
}

// dt = softplus(raw + bias); dA = exp(-exp(A_log)*dt)
__global__ void dt_kernel(const float* __restrict__ zx, const float* __restrict__ dt_bias,
                          const float* __restrict__ A_log,
                          float* __restrict__ dtv, float* __restrict__ dAv) {
    int i = blockIdx.x * 256 + threadIdx.x;
    if (i >= M_TOK * NH) return;
    int hd = i & 7, m = i >> 3;
    float v = zx[(size_t)m * DPROJ + (DIN + CONVCH) + hd] + dt_bias[hd];
    float sp = (v > 20.f) ? v : log1pf(__expf(v));
    dtv[i] = sp;
    dAv[i] = __expf(-__expf(A_log[hd]) * sp);
}

// selective-state scan: one block per (seq, head); thread p owns state row h[p][0..63]
__global__ __launch_bounds__(64) void scan_kernel(
    const float* __restrict__ xbc, const float* __restrict__ dtv,
    const float* __restrict__ dAv, const float* __restrict__ Dp,
    float* __restrict__ y)
{
    int s  = blockIdx.x >> 3;
    int hd = blockIdx.x & 7;
    int p  = threadIdx.x;
    __shared__ float Bsh[DSTATE], Csh[DSTATE];
    float h[DSTATE];
#pragma unroll
    for (int n = 0; n < DSTATE; ++n) h[n] = 0.f;
    float Dh = Dp[hd];
    size_t base = (size_t)s * LSEQ;
    for (int t = 0; t < LSEQ; ++t) {
        size_t m = base + t;
        __syncthreads();
        Bsh[p] = xbc[m * CONVCH + DIN + p];
        Csh[p] = xbc[m * CONVCH + DIN + DSTATE + p];
        __syncthreads();
        float xp = xbc[m * CONVCH + hd * HDIM + p];
        float dA = dAv[m * NH + hd];
        float u  = dtv[m * NH + hd] * xp;
        float acc = 0.f;
#pragma unroll
        for (int n = 0; n < DSTATE; ++n) {
            h[n] = h[n] * dA + u * Bsh[n];
            acc += h[n] * Csh[n];
        }
        y[m * DIN + hd * HDIM + p] = acc + Dh * xp;
    }
}

// y *= silu(z); rmsnorm over 512 with norm_w; write bf16
__global__ __launch_bounds__(256) void gate_norm(
    const float* __restrict__ y, const float* __restrict__ zx,
    const float* __restrict__ norm_w, bf16* __restrict__ out)
{
    int m = blockIdx.x, tid = threadIdx.x;
    __shared__ float sred[256];
    float vals[2];
    float ss = 0.f;
#pragma unroll
    for (int e = 0; e < 2; ++e) {
        int c = tid + e * 256;
        float yv = y[(size_t)m * DIN + c];
        float zv = zx[(size_t)m * DPROJ + c];
        float g = yv * (zv / (1.f + __expf(-zv)));
        vals[e] = g;
        ss += g * g;
    }
    sred[tid] = ss;
    __syncthreads();
    for (int st = 128; st > 0; st >>= 1) {
        if (tid < st) sred[tid] += sred[tid + st];
        __syncthreads();
    }
    float scale = rsqrtf(sred[0] * (1.f / (float)DIN) + 1e-5f);
#pragma unroll
    for (int e = 0; e < 2; ++e) {
        int c = tid + e * 256;
        out[(size_t)m * DIN + c] = f2bf(vals[e] * scale * norm_w[c]);
    }
}

extern "C" void kernel_launch(void* const* d_in, const int* in_sizes, int n_in,
                              void* d_out, int out_size, void* d_ws, size_t ws_size,
                              hipStream_t stream) {
    (void)in_sizes; (void)n_in; (void)out_size; (void)ws_size;
    const float* x = (const float*)d_in[0];
    const float* in_w[2]    = {(const float*)d_in[1],  (const float*)d_in[9]};
    const float* conv_w[2]  = {(const float*)d_in[2],  (const float*)d_in[10]};
    const float* conv_b[2]  = {(const float*)d_in[3],  (const float*)d_in[11]};
    const float* dt_bias[2] = {(const float*)d_in[4],  (const float*)d_in[12]};
    const float* A_log[2]   = {(const float*)d_in[5],  (const float*)d_in[13]};
    const float* Dp[2]      = {(const float*)d_in[6],  (const float*)d_in[14]};
    const float* norm_w[2]  = {(const float*)d_in[7],  (const float*)d_in[15]};
    const float* out_w[2]   = {(const float*)d_in[8],  (const float*)d_in[16]};
    const float* fc_w = (const float*)d_in[17];
    const float* fc_b = (const float*)d_in[18];

    size_t off = 0;
    auto take = [&](size_t bytes) -> char* {
        off = (off + 255) & ~(size_t)255;
        char* p = (char*)d_ws + off;
        off += bytes;
        return p;
    };
    bf16* wbf_in[2], *wbf_out[2];
    float* zx[2], *xbc[2], *dtv[2], *dAv[2], *ybuf[2];
    bf16* ynorm[2];
    for (int d = 0; d < 2; ++d) {
        wbf_in[d]  = (bf16*)take((size_t)NPADIN * DM * 2);
        wbf_out[d] = (bf16*)take((size_t)DM * DIN * 2);
        zx[d]   = (float*)take((size_t)M_TOK * DPROJ * 4);
        xbc[d]  = (float*)take((size_t)M_TOK * CONVCH * 4);
        dtv[d]  = (float*)take((size_t)M_TOK * NH * 4);
        dAv[d]  = (float*)take((size_t)M_TOK * NH * 4);
        ybuf[d] = (float*)take((size_t)M_TOK * DIN * 4);
        ynorm[d]= (bf16*)take((size_t)M_TOK * DIN * 2);
    }
    bf16* wcat  = (bf16*)take((size_t)DM * DIN * 2);
    bf16* xh_bf = (bf16*)take((size_t)M_TOK * DM * 2);
    bf16* xv_bf = (bf16*)take((size_t)M_TOK * DM * 2);
    bf16* acat  = (bf16*)take((size_t)M_TOK * DIN * 2);

    // ---- weight / activation prep ----
    for (int d = 0; d < 2; ++d) {
        cvt_pad_bf16<<<(NPADIN * DM + 255) / 256, 256, 0, stream>>>(
            in_w[d], wbf_in[d], DPROJ * DM, NPADIN * DM);
        cvt_pad_bf16<<<(DM * DIN + 255) / 256, 256, 0, stream>>>(
            out_w[d], wbf_out[d], DM * DIN, DM * DIN);
    }
    build_wcat<<<(DM * DIN + 255) / 256, 256, 0, stream>>>(fc_w, wcat);
    cvt_pad_bf16<<<(M_TOK * DM + 255) / 256, 256, 0, stream>>>(
        x, xh_bf, M_TOK * DM, M_TOK * DM);
    cvt_xv<<<(M_TOK * DM + 255) / 256, 256, 0, stream>>>(x, xv_bf);

    // ---- two mamba passes (d=0: horizontal / mh, d=1: vertical / mv) ----
    for (int d = 0; d < 2; ++d) {
        const bf16* xin = d ? xv_bf : xh_bf;
        dim3 gIn(M_TOK / 64, NPADIN / 64);
        gemm_bf16_wmma<<<gIn, 256, 0, stream>>>(
            xin, DM, wbf_in[d], DM, zx[d], DPROJ, DPROJ, DM, 0, 0, nullptr);

        conv_silu<<<NSEQ, 256, 0, stream>>>(zx[d], conv_w[d], conv_b[d], xbc[d]);
        dt_kernel<<<(M_TOK * NH + 255) / 256, 256, 0, stream>>>(
            zx[d], dt_bias[d], A_log[d], dtv[d], dAv[d]);
        scan_kernel<<<NSEQ * NH, 64, 0, stream>>>(xbc[d], dtv[d], dAv[d], Dp[d], ybuf[d]);
        gate_norm<<<M_TOK, 256, 0, stream>>>(ybuf[d], zx[d], norm_w[d], ynorm[d]);

        // out-proj into concatenated fc activations:
        //   d=0 (h1) -> cols 256..511, identity rows
        //   d=1 (v1) -> cols 0..255, rows permuted (b,w,h)->(b,h,w)
        int flags = 1 | (d ? 2 : 0);
        int colOff = d ? 0 : 256;
        dim3 gOut(M_TOK / 64, DM / 64);
        gemm_bf16_wmma<<<gOut, 256, 0, stream>>>(
            ynorm[d], DIN, wbf_out[d], DIN, acat, DIN, DM, DIN, flags, colOff, nullptr);
    }

    // ---- fused fc: out = acat @ wcat^T + fc_b ----
    dim3 gFc(M_TOK / 64, DM / 64);
    gemm_bf16_wmma<<<gFc, 256, 0, stream>>>(
        acat, DIN, wcat, DIN, (float*)d_out, DM, DM, DIN, 4, 0, fc_b);
}
